// GNN_63290638074640
// MI455X (gfx1250) — compile-verified, matrix operand order checked by reading, not run on hardware
//
#include <hip/hip_runtime.h>
#include <hip/hip_bf16.h>

// ---------------------------------------------------------------------------
// Problem constants (from reference)
// ---------------------------------------------------------------------------
#define G_    512
#define NMAX  256
#define E_    1024
#define F_IN  30
#define KPAD  32            // F_IN padded to WMMA K granularity
#define EMB   256
#define H_    3
#define HD    768           // H*EMB
#define NCLS  2
#define K1_   205
#define K2_   164
#define K3_   132
#define NEG_SLOPE 0.2f

typedef __attribute__((ext_vector_type(16))) __bf16 v16bf;
typedef __attribute__((ext_vector_type(8)))  __bf16 v8bf;
typedef __attribute__((ext_vector_type(8)))  float  v8f;
typedef __attribute__((ext_vector_type(4)))  unsigned u32x4;
typedef __attribute__((ext_vector_type(8)))  int      i32x8;
typedef __attribute__((ext_vector_type(4)))  int      i32x4;

#if __has_builtin(__builtin_amdgcn_tensor_load_to_lds) && \
    __has_builtin(__builtin_amdgcn_s_wait_tensorcnt)
#define HAVE_TDM 1
#else
#define HAVE_TDM 0
#endif

// ---------------------------------------------------------------------------
// Batched GEMM, bf16 WMMA, f32 accumulate:
//   C[b,m,n] = act( sum_k A[b,m,k] * W[n,k] + bias[n] )
// A, W pre-converted to bf16 in global memory.
// REQUIRES: M % 128 == 0, N % 64 == 0, K % 32 == 0 (caller pads).
// Block = 256 threads = 8 waves (4 M x 2 N); wave owns 32x32 -> 4 v_wmma/step.
// Tiles staged into LDS by the Tensor Data Mover (TDM) with hardware padding
// that reproduces the LDT=40-half skewed row layout; fragments are then two
// contiguous ds_load_b128 per the CDNA5 16-bit lane->K mapping:
//   element e<8 : K = e + 8*lhalf ; e>=8 : K = e + 8 + 8*lhalf
// ---------------------------------------------------------------------------
#define BM 128
#define BN 64
#define BK 32
#define LDT 40   // LDS row stride in halfs (64B row + 16B pad)

__device__ __forceinline__ v16bf make_frag(const __bf16* row, int lhalf)
{
    v8bf lo = *(const v8bf*)(row + 8 * lhalf);
    v8bf hi = *(const v8bf*)(row + 16 + 8 * lhalf);
    return __builtin_shufflevector(lo, hi, 0, 1, 2, 3, 4, 5, 6, 7,
                                           8, 9, 10, 11, 12, 13, 14, 15);
}

__device__ __forceinline__ v8bf cvt8(float4 v0, float4 v1)
{
    v8bf h;
    h[0] = (__bf16)v0.x; h[1] = (__bf16)v0.y; h[2] = (__bf16)v0.z; h[3] = (__bf16)v0.w;
    h[4] = (__bf16)v1.x; h[5] = (__bf16)v1.y; h[6] = (__bf16)v1.z; h[7] = (__bf16)v1.w;
    return h;
}

#if HAVE_TDM
// Issue one 2D TDM tile load: tileH rows x 32 bf16 from row-major [.., rowLen]
// global data into LDS at ldsOff, with 16B padding after each 64B row.
__device__ __forceinline__ void tdm_load_2d(unsigned ldsOff, const __bf16* gbase,
                                            long elemOff, int rowLen, int tileH)
{
    unsigned long long ga = (unsigned long long)gbase + (unsigned long long)elemOff * 2ull;
    u32x4 g0;
    g0[0] = 1u;                                   // count=1 (valid), user mode
    g0[1] = ldsOff;                               // lds_addr (bytes)
    g0[2] = (unsigned)(ga & 0xFFFFFFFFull);       // global_addr[31:0]
    g0[3] = (unsigned)((ga >> 32) & 0x01FFFFFFull) | 0x80000000u; // addr hi | type=2
    i32x8 g1;
    // data_size=2B(code1), pad_enable, pad_interval=3 (16 DWORDs), pad_amount=3 (4 DWORDs)
    g1[0] = (1 << 16) | (1 << 20) | (3 << 22) | (3 << 25);
    g1[1] = (int)(((unsigned)rowLen & 0xFFFFu) << 16);              // tensor_dim0 lo16
    g1[2] = (int)(((unsigned)rowLen >> 16) | ((unsigned)tileH << 16)); // dim0 hi | dim1 lo
    g1[3] = (int)(BK << 16);                                        // dim1 hi(0) | tile_dim0
    g1[4] = tileH;                                                  // tile_dim1 | tile_dim2=0
    g1[5] = rowLen;                                                 // tensor_dim0_stride lo32
    g1[6] = 0;
    g1[7] = 0;
    i32x4 z4 = {0, 0, 0, 0};
#if defined(__clang_major__) && (__clang_major__ >= 23)
    i32x8 z8 = {0, 0, 0, 0, 0, 0, 0, 0};
    __builtin_amdgcn_tensor_load_to_lds(g0, g1, z4, z4, z8, 0);
#else
    __builtin_amdgcn_tensor_load_to_lds(g0, g1, z4, z4, 0);
#endif
}
#endif

__global__ __launch_bounds__(256)
void gemm_wmma_bf16(const __bf16* __restrict__ A, const __bf16* __restrict__ Wt,
                    const float* __restrict__ bias, float* __restrict__ C,
                    int K, int N, long aBatch, long cBatch, int act)
{
    __shared__ __attribute__((aligned(16))) __bf16 As[BM * LDT];
    __shared__ __attribute__((aligned(16))) __bf16 Bs[BN * LDT];

    const int g = blockIdx.z;
    A += (long)g * aBatch;
    C += (long)g * cBatch;

    const int tid   = threadIdx.x;
    const int wave  = tid >> 5;
    const int lane  = tid & 31;
    const int wm    = wave & 3;
    const int wn    = wave >> 2;
    const int lhalf = lane >> 4;
    const int lmod  = lane & 15;

    const int m0 = blockIdx.y * BM;
    const int n0 = blockIdx.x * BN;

    v8f acc00 = {}, acc01 = {}, acc10 = {}, acc11 = {};

#if HAVE_TDM
    const unsigned ldsA = (unsigned)(unsigned long long)(void*)&As[0];
    const unsigned ldsB = (unsigned)(unsigned long long)(void*)&Bs[0];
#else
    const int ra0 = (tid + 0)   >> 2, qa0 = (tid + 0)   & 3;
    const int ra1 = (tid + 256) >> 2, qa1 = (tid + 256) & 3;
    const int rb  = tid >> 2,         qb  = tid & 3;
#endif

    for (int k0 = 0; k0 < K; k0 += BK) {
#if HAVE_TDM
        if (wave == 0) {
            tdm_load_2d(ldsA, A,  (long)m0 * K + k0, K, BM);
            tdm_load_2d(ldsB, Wt, (long)n0 * K + k0, K, BN);
            __builtin_amdgcn_s_wait_tensorcnt(0);
        }
        __syncthreads();
#else
        // Fallback: plain 16B global->LDS copies (data already bf16)
        *(uint4*)&As[ra0 * LDT + qa0 * 8] =
            *(const uint4*)(A + (long)(m0 + ra0) * K + k0 + qa0 * 8);
        *(uint4*)&As[ra1 * LDT + qa1 * 8] =
            *(const uint4*)(A + (long)(m0 + ra1) * K + k0 + qa1 * 8);
        *(uint4*)&Bs[rb * LDT + qb * 8] =
            *(const uint4*)(Wt + (long)(n0 + rb) * K + k0 + qb * 8);
        __syncthreads();
#endif

        v16bf a0 = make_frag(&As[(wm * 32 +      lmod) * LDT], lhalf);
        v16bf a1 = make_frag(&As[(wm * 32 + 16 + lmod) * LDT], lhalf);
        v16bf b0 = make_frag(&Bs[(wn * 32 +      lmod) * LDT], lhalf);
        v16bf b1 = make_frag(&Bs[(wn * 32 + 16 + lmod) * LDT], lhalf);

        acc00 = __builtin_amdgcn_wmma_f32_16x16x32_bf16(false, a0, false, b0,
                                                        (short)0, acc00, false, false);
        acc01 = __builtin_amdgcn_wmma_f32_16x16x32_bf16(false, a0, false, b1,
                                                        (short)0, acc01, false, false);
        acc10 = __builtin_amdgcn_wmma_f32_16x16x32_bf16(false, a1, false, b0,
                                                        (short)0, acc10, false, false);
        acc11 = __builtin_amdgcn_wmma_f32_16x16x32_bf16(false, a1, false, b1,
                                                        (short)0, acc11, false, false);
        __syncthreads();
    }

    const int mBase = m0 + wm * 32 + 8 * lhalf;
    const int nBase = n0 + wn * 32 + lmod;
#pragma unroll
    for (int v = 0; v < 8; ++v) {
        float b0v = bias ? bias[nBase] : 0.0f;
        float b1v = bias ? bias[nBase + 16] : 0.0f;
        float v00 = acc00[v] + b0v, v01 = acc01[v] + b1v;
        float v10 = acc10[v] + b0v, v11 = acc11[v] + b1v;
        if (act == 1) {
            v00 = fmaxf(v00, 0.f); v01 = fmaxf(v01, 0.f);
            v10 = fmaxf(v10, 0.f); v11 = fmaxf(v11, 0.f);
        }
        C[(long)(mBase + v)      * N + nBase]      = v00;
        C[(long)(mBase + v)      * N + nBase + 16] = v01;
        C[(long)(mBase + 16 + v) * N + nBase]      = v10;
        C[(long)(mBase + 16 + v) * N + nBase + 16] = v11;
    }
}

// ---------------------------------------------------------------------------
// Ordered-uint encoding so unsigned atomicMax == float max
// ---------------------------------------------------------------------------
__device__ __forceinline__ unsigned f2ord(float f) {
    unsigned u = __float_as_uint(f);
    return (u & 0x80000000u) ? ~u : (u | 0x80000000u);
}
__device__ __forceinline__ float ord2f(unsigned u) {
    return __uint_as_float((u & 0x80000000u) ? (u & 0x7FFFFFFFu) : ~u);
}

// ---------------------------------------------------------------------------
// GAT attention + aggregation. One block (256 threads) per graph.
// ---------------------------------------------------------------------------
__global__ __launch_bounds__(256)
void gat_attention(const float* __restrict__ h,
                   const int* __restrict__ src, const int* __restrict__ dst,
                   const int* __restrict__ mask,
                   const float* __restrict__ asrc, const float* __restrict__ adst,
                   float* __restrict__ out, int n, int e)
{
    __shared__ float    s_as[NMAX * H_];
    __shared__ float    s_ad[NMAX * H_];
    __shared__ unsigned s_mx[NMAX * H_];
    __shared__ float    s_den[NMAX * H_];

    const int g   = blockIdx.x;
    const int tid = threadIdx.x;
    const float* hg = h + (long)g * NMAX * HD;
    float*       og = out + (long)g * NMAX * HD;
    const int*   sg = src + (long)g * E_;
    const int*   dg = dst + (long)g * E_;
    const int*   mg = mask ? (mask + (long)g * E_) : nullptr;

    for (int i = tid; i < n * H_; i += 256) {
        int node = i / H_, hh = i % H_;
        const float* hv = hg + (long)node * HD + hh * EMB;
        const float* av = asrc + hh * EMB;
        const float* bv = adst + hh * EMB;
        float ss = 0.f, dd = 0.f;
        for (int c = 0; c < EMB; ++c) { float v = hv[c]; ss += v * av[c]; dd += v * bv[c]; }
        s_as[i] = ss; s_ad[i] = dd;
    }
    for (int i = tid; i < NMAX * H_; i += 256) { s_mx[i] = 0u; s_den[i] = 0.f; }
    __syncthreads();

    const int total = e + n;

    for (int t = tid; t < total; t += 256) {
        int s, d, valid;
        if (t < e) { s = sg[t]; d = dg[t]; valid = mg ? mg[t] : 1; }
        else       { s = d = t - e; valid = 1; }
        if (!valid) continue;
        for (int hh = 0; hh < H_; ++hh) {
            float l = s_as[s * H_ + hh] + s_ad[d * H_ + hh];
            l = (l > 0.f) ? l : NEG_SLOPE * l;
            atomicMax(&s_mx[d * H_ + hh], f2ord(l));
        }
    }
    __syncthreads();

    for (int t = tid; t < total; t += 256) {
        int s, d, valid;
        if (t < e) { s = sg[t]; d = dg[t]; valid = mg ? mg[t] : 1; }
        else       { s = d = t - e; valid = 1; }
        if (!valid) continue;
        for (int hh = 0; hh < H_; ++hh) {
            float l = s_as[s * H_ + hh] + s_ad[d * H_ + hh];
            l = (l > 0.f) ? l : NEG_SLOPE * l;
            atomicAdd(&s_den[d * H_ + hh], __expf(l - ord2f(s_mx[d * H_ + hh])));
        }
    }
    __syncthreads();

    for (int t = tid; t < total; t += 256) {
        int s, d, valid;
        if (t < e) { s = sg[t]; d = dg[t]; valid = mg ? mg[t] : 1; }
        else       { s = d = t - e; valid = 1; }
        if (!valid) continue;
        for (int hh = 0; hh < H_; ++hh) {
            float l = s_as[s * H_ + hh] + s_ad[d * H_ + hh];
            l = (l > 0.f) ? l : NEG_SLOPE * l;
            float alpha = __expf(l - ord2f(s_mx[d * H_ + hh])) / s_den[d * H_ + hh];
            const float* hs = hg + (long)s * HD + hh * EMB;
            float*       od = og + (long)d * HD + hh * EMB;
            for (int c = 0; c < EMB; ++c) atomicAdd(&od[c], alpha * hs[c]);
        }
    }
}

// ---------------------------------------------------------------------------
// TopK pooling: one block per graph, bitonic sort in LDS.
// ---------------------------------------------------------------------------
__global__ __launch_bounds__(256)
void topk_pool(const float* __restrict__ x, const float* __restrict__ p,
               const int* __restrict__ srcIn, const int* __restrict__ dstIn,
               const int* __restrict__ maskIn,
               float* __restrict__ xOut, int* __restrict__ srcOut,
               int* __restrict__ dstOut, int* __restrict__ maskOut,
               int n, int k)
{
    __shared__ float s_score[NMAX];
    __shared__ int   s_idx[NMAX];
    __shared__ int   s_inv[NMAX];

    const int g   = blockIdx.x;
    const int tid = threadIdx.x;
    const float* xg = x + (long)g * NMAX * EMB;
    float*       yg = xOut + (long)g * NMAX * EMB;

    float pn = 0.f;
    for (int c = 0; c < EMB; ++c) pn += p[c] * p[c];
    pn = sqrtf(pn);

    float sc = -1e30f;
    if (tid < n) {
        const float* xi = xg + (long)tid * EMB;
        float acc = 0.f;
        for (int c = 0; c < EMB; ++c) acc += xi[c] * p[c];
        sc = tanhf(acc / pn);
    }
    s_score[tid] = sc;
    s_idx[tid]   = tid;
    __syncthreads();

    for (int size = 2; size <= NMAX; size <<= 1) {
        for (int stride = size >> 1; stride > 0; stride >>= 1) {
            int i = tid, j = i ^ stride;
            if (j > i) {
                float si = s_score[i], sj = s_score[j];
                int   ii = s_idx[i],  jj = s_idx[j];
                bool before = (si > sj) || (si == sj && ii < jj);
                bool desc   = ((i & size) == 0);
                if (desc ? !before : before) {
                    s_score[i] = sj; s_score[j] = si;
                    s_idx[i]   = jj; s_idx[j]   = ii;
                }
            }
            __syncthreads();
        }
    }

    s_inv[tid] = -1;
    __syncthreads();
    if (tid < k) s_inv[s_idx[tid]] = tid;
    __syncthreads();

    for (int j = 0; j < NMAX; ++j) {
        float v = 0.f;
        if (j < k) v = xg[(long)s_idx[j] * EMB + tid] * s_score[j];
        yg[(long)j * EMB + tid] = v;
    }

    const int* sI = srcIn  + (long)g * E_;
    const int* dI = dstIn  + (long)g * E_;
    const int* mI = maskIn ? (maskIn + (long)g * E_) : nullptr;
    int* sO = srcOut  + (long)g * E_;
    int* dO = dstOut  + (long)g * E_;
    int* mO = maskOut + (long)g * E_;
    for (int t = tid; t < E_; t += 256) {
        int ns = s_inv[sI[t]], nd = s_inv[dI[t]];
        int valid = (mI ? mI[t] : 1) && (ns >= 0) && (nd >= 0);
        sO[t] = ns > 0 ? ns : 0;
        dO[t] = nd > 0 ? nd : 0;
        mO[t] = valid;
    }
}

// ---------------------------------------------------------------------------
// Readout (max || mean over k rows) accumulated into z [G, 2*EMB]
// ---------------------------------------------------------------------------
__global__ __launch_bounds__(256)
void readout_add(const float* __restrict__ x, float* __restrict__ z, int k)
{
    const int g = blockIdx.x, c = threadIdx.x;
    const float* xg = x + (long)g * NMAX * EMB;
    float mx = -1e30f, sm = 0.f;
    for (int j = 0; j < k; ++j) {
        float v = xg[(long)j * EMB + c];
        mx = fmaxf(mx, v);
        sm += v;
    }
    z[(long)g * 2 * EMB + c]        += mx;
    z[(long)g * 2 * EMB + EMB + c] += sm / (float)k;
}

// ---------------------------------------------------------------------------
// Utility kernels
// ---------------------------------------------------------------------------
__global__ void zero_rows(float* buf, long batchStride, long count)
{
    float* pB = buf + (long)blockIdx.x * batchStride;
    for (long i = threadIdx.x; i < count; i += blockDim.x) pB[i] = 0.f;
}

__global__ void add_row_bias(float* buf, const float* __restrict__ bias,
                             long batchStride, long count, int cols)
{
    float* pB = buf + (long)blockIdx.x * batchStride;
    for (long i = threadIdx.x; i < count; i += blockDim.x)
        pB[i] += bias[i % cols];
}

// Vectorized fp32 -> bf16 (n8 = element count / 8)
__global__ void convert_bf16_vec(__bf16* __restrict__ dst, const float* __restrict__ src,
                                 long n8)
{
    long stride = (long)gridDim.x * blockDim.x;
    for (long i = (long)blockIdx.x * blockDim.x + threadIdx.x; i < n8; i += stride) {
        const float4* s = (const float4*)(src + i * 8);
        float4 v0 = s[0], v1 = s[1];
        *(v8bf*)(dst + i * 8) = cvt8(v0, v1);
    }
}

// Pad rows kIn -> kOut with zeros, fp32 -> bf16
__global__ void pad_convert_bf16(__bf16* __restrict__ dst, const float* __restrict__ src,
                                 long rows, int kIn, int kOut)
{
    long total = rows * kOut;
    long stride = (long)gridDim.x * blockDim.x;
    for (long i = (long)blockIdx.x * blockDim.x + threadIdx.x; i < total; i += stride) {
        long r = i / kOut; int c = (int)(i % kOut);
        dst[i] = (c < kIn) ? (__bf16)src[r * kIn + c] : (__bf16)0.f;
    }
}

__global__ void lin2_kernel(const float* __restrict__ zz, const float* __restrict__ w,
                            const float* __restrict__ b, float* __restrict__ out)
{
    int idx = blockIdx.x * blockDim.x + threadIdx.x;
    if (idx >= G_ * NCLS) return;
    int g = idx / NCLS, c = idx % NCLS;
    float acc = b[c];
    const float* zr = zz + (long)g * EMB;
    const float* wr = w + (long)c * EMB;
    for (int k = 0; k < EMB; ++k) acc += zr[k] * wr[k];
    out[idx] = acc;
}

// ---------------------------------------------------------------------------
// Host orchestration
// ---------------------------------------------------------------------------
extern "C" void kernel_launch(void* const* d_in, const int* in_sizes, int n_in,
                              void* d_out, int out_size, void* d_ws, size_t ws_size,
                              hipStream_t stream)
{
    (void)in_sizes; (void)n_in; (void)out_size; (void)ws_size;

    const float* x    = (const float*)d_in[0];
    const int*   esrc = (const int*)d_in[1];
    const int*   edst = (const int*)d_in[2];
    const float* Wl[3]   = {(const float*)d_in[3],  (const float*)d_in[10], (const float*)d_in[17]};
    const float* As_[3]  = {(const float*)d_in[4],  (const float*)d_in[11], (const float*)d_in[18]};
    const float* Ad_[3]  = {(const float*)d_in[5],  (const float*)d_in[12], (const float*)d_in[19]};
    const float* Bb[3]   = {(const float*)d_in[6],  (const float*)d_in[13], (const float*)d_in[20]};
    const float* Htw[3]  = {(const float*)d_in[7],  (const float*)d_in[14], (const float*)d_in[21]};
    const float* Htb[3]  = {(const float*)d_in[8],  (const float*)d_in[15], (const float*)d_in[22]};
    const float* Pp[3]   = {(const float*)d_in[9],  (const float*)d_in[16], (const float*)d_in[23]};
    const float* l1w = (const float*)d_in[24];
    const float* l1b = (const float*)d_in[25];
    const float* l2w = (const float*)d_in[26];
    const float* l2b = (const float*)d_in[27];

    // ---------------- Workspace layout ----------------
    float* wsf = (float*)d_ws;
    size_t off = 0;
    float* bufA = wsf + off; off += (size_t)G_ * NMAX * HD;    // pre-agg h (fp32)
    float* bufB = wsf + off; off += (size_t)G_ * NMAX * HD;    // agg out (fp32)
    float* bufC = wsf + off; off += (size_t)G_ * NMAX * EMB;   // post-ht h (fp32)
    float* bufD = wsf + off; off += (size_t)G_ * NMAX * EMB;   // pooled x (fp32)
    float* zbuf = wsf + off; off += (size_t)G_ * 2 * EMB;      // readout accumulator
    int* ib = (int*)(wsf + off);
    const size_t GE = (size_t)G_ * E_;
    int* srcA = ib;            int* dstA = ib + GE;     int* mskA = ib + 2 * GE;
    int* srcB = ib + 3 * GE;   int* dstB = ib + 4 * GE; int* mskB = ib + 5 * GE;
    // bf16 region (16B-aligned: all preceding counts are multiples of 8)
    __bf16* hb = (__bf16*)(ib + 6 * GE);
    size_t hoff = 0;
    __bf16* xbf   = hb + hoff; hoff += (size_t)G_ * NMAX * KPAD; // padded x
    __bf16* w1bf  = hb + hoff; hoff += (size_t)HD * KPAD;        // padded W1
    __bf16* w2bf  = hb + hoff; hoff += (size_t)HD * EMB;
    __bf16* w3bf  = hb + hoff; hoff += (size_t)HD * EMB;
    __bf16* htwbf = hb + hoff; hoff += (size_t)3 * EMB * HD;     // 3 ht weights
    __bf16* l1wbf = hb + hoff; hoff += (size_t)EMB * 2 * EMB;
    __bf16* bbf   = hb + hoff; hoff += (size_t)G_ * NMAX * HD;   // bf16 of bufB
    __bf16* dbf   = hb + hoff; hoff += (size_t)G_ * NMAX * EMB;  // bf16 of bufD
    __bf16* zbf   = hb + hoff; hoff += (size_t)G_ * 2 * EMB;     // bf16 of z

    // ---------------- Weight / input conversion (bf16) ----------------
    pad_convert_bf16<<<2048, 256, 0, stream>>>(xbf, x, (long)G_ * NMAX, F_IN, KPAD);
    pad_convert_bf16<<<32, 256, 0, stream>>>(w1bf, Wl[0], HD, F_IN, KPAD);
    convert_bf16_vec<<<96, 256, 0, stream>>>(w2bf, Wl[1], (long)HD * EMB / 8);
    convert_bf16_vec<<<96, 256, 0, stream>>>(w3bf, Wl[2], (long)HD * EMB / 8);
    for (int s = 0; s < 3; ++s)
        convert_bf16_vec<<<96, 256, 0, stream>>>(htwbf + (size_t)s * EMB * HD, Htw[s],
                                                 (long)EMB * HD / 8);
    convert_bf16_vec<<<64, 256, 0, stream>>>(l1wbf, l1w, (long)EMB * 2 * EMB / 8);

    // z = 0
    zero_rows<<<G_, 256, 0, stream>>>(zbuf, 2 * EMB, 2 * EMB);

    const int nIn[3]  = {NMAX, K1_, K2_};
    const int nOut[3] = {K1_, K2_, K3_};

    for (int s = 0; s < 3; ++s) {
        const __bf16* xin  = (s == 0) ? xbf : dbf;
        const long    xinB = (s == 0) ? (long)NMAX * KPAD : (long)NMAX * EMB;
        const int     dInF = (s == 0) ? KPAD : EMB;
        const __bf16* Wg   = (s == 0) ? w1bf : (s == 1 ? w2bf : w3bf);
        const int* sIn = (s == 0) ? esrc : (s == 1 ? srcA : srcB);
        const int* dIn = (s == 0) ? edst : (s == 1 ? dstA : dstB);
        const int* mIn = (s == 0) ? nullptr : (s == 1 ? mskA : mskB);
        int* sOut = (s == 1) ? srcB : srcA;
        int* dOut = (s == 1) ? dstB : dstA;
        int* mOut = (s == 1) ? mskB : mskA;
        const int ni = nIn[s], no = nOut[s];

        // 1) bufA = xin @ W.T  (full NMAX rows; rows >= ni unread)
        gemm_wmma_bf16<<<dim3(HD / BN, NMAX / BM, G_), 256, 0, stream>>>(
            xin, Wg, nullptr, bufA, dInF, HD, xinB, (long)NMAX * HD, 0);

        // 2) zero aggregation target
        zero_rows<<<G_, 256, 0, stream>>>(bufB, (long)NMAX * HD, (long)NMAX * HD);

        // 3) attention softmax + aggregation
        gat_attention<<<G_, 256, 0, stream>>>(bufA, sIn, dIn, mIn, As_[s], Ad_[s],
                                              bufB, ni, E_);

        // 4) + b (GAT bias) on the live rows, then convert to bf16
        add_row_bias<<<G_, 256, 0, stream>>>(bufB, Bb[s], (long)NMAX * HD,
                                             (long)ni * HD, HD);
        convert_bf16_vec<<<2048, 256, 0, stream>>>(bbf, bufB, (long)G_ * NMAX * HD / 8);

        // 5) bufC = bufB @ ht_w.T + ht_b
        gemm_wmma_bf16<<<dim3(EMB / BN, NMAX / BM, G_), 256, 0, stream>>>(
            bbf, htwbf + (size_t)s * EMB * HD, Htb[s], bufC,
            HD, EMB, (long)NMAX * HD, (long)NMAX * EMB, 0);

        // 6) TopK pooling -> bufD (+ bf16 copy for next stage's GEMM)
        topk_pool<<<G_, 256, 0, stream>>>(bufC, Pp[s], sIn, dIn, mIn,
                                          bufD, sOut, dOut, mOut, ni, no);
        convert_bf16_vec<<<1024, 256, 0, stream>>>(dbf, bufD, (long)G_ * NMAX * EMB / 8);

        // 7) z += readout(bufD)
        readout_add<<<G_, 256, 0, stream>>>(bufD, zbuf, no);
    }

    // lin1: zz = relu(z @ l1w.T + l1b)
    convert_bf16_vec<<<64, 256, 0, stream>>>(zbf, zbuf, (long)G_ * 2 * EMB / 8);
    gemm_wmma_bf16<<<dim3(EMB / BN, G_ / BM, 1), 256, 0, stream>>>(
        zbf, l1wbf, l1b, bufA, 2 * EMB, EMB, 0, 0, 1);

    // lin2: out = zz @ l2w.T + l2b
    lin2_kernel<<<(G_ * NCLS + 255) / 256, 256, 0, stream>>>(bufA, l2w, l2b, (float*)d_out);
}
